// SegonaCapa_91010357002914
// MI455X (gfx1250) — compile-verified, matrix operand order checked by reading, not run on hardware
//
#include <hip/hip_runtime.h>
#include <hip/hip_bf16.h>

// MI455X (gfx1250) implementation of the video-UNet forward pass.
// All GEMM-shaped work (3x3/1x1/4x4-stride2 convs, transposed convs via
// parity decomposition, attention projections) runs through a 64x64-tile
// implicit-GEMM using v_wmma_f32_16x16x32_f16 (fp32 storage, f16 tiles,
// f32 accumulate). Staging is branch-free (clamped loads + cndmask),
// LDS traffic is packed (ds_store_b32 / ds_load_b128), epilogues are
// branch-free via always-valid row/col bias pointers.

typedef _Float16 f16;
typedef __attribute__((ext_vector_type(8)))  _Float16 v8h;
typedef __attribute__((ext_vector_type(16))) _Float16 v16h;
typedef __attribute__((ext_vector_type(8)))  float    v8f;

static inline int cdiv_h(int a, int b) { return (a + b - 1) / b; }

__device__ __forceinline__ float siluf(float v) { return v * (1.0f / (1.0f + __expf(-v))); }

__device__ __forceinline__ v8f zero8() {
  v8f z;
#pragma unroll
  for (int e = 0; e < 8; ++e) z[e] = 0.0f;
  return z;
}

__device__ __forceinline__ v8f wmma16x16x32(v16h a, v16h b, v8f c) {
  return __builtin_amdgcn_wmma_f32_16x16x32_f16(false, a, false, b, (short)0, c, false, false);
}

// Pack two fp32 into one dword of two f16.
__device__ __forceinline__ unsigned pack2h(float x, float y) {
  union { f16 h[2]; unsigned u; } un;
  un.h[0] = (f16)x;
  un.h[1] = (f16)y;
  return un.u;
}

// Both A and B(transposed) tiles live in LDS as [64 rows][32 K] f16, row-major.
// ISA 16-bit operand layout: lanes 0-15 -> K {0..7,16..23}; lanes 16-31 -> {8..15,24..31}.
// kb = (lane>=16)*8; fragment = two contiguous 8xf16 chunks -> 2x ds_load_b128.
__device__ __forceinline__ v16h frag_row(const f16* row, int kb) {
  const v8h lo = *(const v8h*)(row + kb);
  const v8h hi = *(const v8h*)(row + 16 + kb);
  return __builtin_shufflevector(lo, hi, 0, 1, 2, 3, 4, 5, 6, 7,
                                 8, 9, 10, 11, 12, 13, 14, 15);
}

// ---------------------------------------------------------------------------
// Generic GEMM: C[M,N] = A[M,K] * B[K,N] + bias_r[m] + bias_c[n].
// bias_r / bias_c are ALWAYS valid pointers (unused one -> zero scratch).
// Row-major, element strides; grid.z batches via sA/sB/sC. K must be a
// multiple of 32 (true for every call site in this network).
// ---------------------------------------------------------------------------
__global__ void sc_gemm_wmma(const float* __restrict__ A, long lda, long sA,
                             const float* __restrict__ B, long ldb, long sB,
                             float* __restrict__ Co, long ldc, long sC,
                             const float* __restrict__ bias_r,
                             const float* __restrict__ bias_c,
                             int M, int N, int K) {
  __shared__ __align__(16) f16 As[64 * 32];
  __shared__ __align__(16) f16 Bs[64 * 32];  // transposed: [col][K]
  const int tid = threadIdx.x, lane = tid & 31, wave = tid >> 5;
  const int mblk = blockIdx.y * 64, nblk = blockIdx.x * 64;
  const int m0 = (wave & 3) * 16, n0 = (wave >> 2) * 32;
  A  += (long)blockIdx.z * sA;
  B  += (long)blockIdx.z * sB;
  Co += (long)blockIdx.z * sC;
  v8f acc0 = zero8(), acc1 = zero8();
  for (int k0 = 0; k0 < K; k0 += 32) {
    float2 av[4];
    float  bv[4][2];
    const int kn = (k0 + 32 < K) ? (k0 + 32) : k0;  // clamped next slab (prefetch)
#pragma unroll
    for (int t = 0; t < 4; ++t) {
      const int pi = tid + t * 256;
      const int r = pi >> 4, kp = (pi & 15) << 1;
      const int gm = mblk + r;
      const int gmc = gm < M ? gm : M - 1;
      const float2 val = *(const float2*)(A + (long)gmc * lda + (k0 + kp));
      av[t].x = gm < M ? val.x : 0.0f;
      av[t].y = gm < M ? val.y : 0.0f;
      __builtin_prefetch(A + (long)gmc * lda + (kn + kp), 0, 1);
    }
#pragma unroll
    for (int t = 0; t < 4; ++t) {
      const int pi = tid + t * 256;
      const int c = pi & 63, kp = (pi >> 6) << 1;
      const int gn = nblk + c;
      const int gnc = gn < N ? gn : N - 1;
      const float b0 = B[(long)(k0 + kp) * ldb + gnc];
      const float b1 = B[(long)(k0 + kp + 1) * ldb + gnc];
      bv[t][0] = gn < N ? b0 : 0.0f;
      bv[t][1] = gn < N ? b1 : 0.0f;
      __builtin_prefetch(B + (long)(kn + kp) * ldb + gnc, 0, 1);
    }
    __syncthreads();  // previous tiles fully consumed
#pragma unroll
    for (int t = 0; t < 4; ++t) {
      const int pi = tid + t * 256;
      const int r = pi >> 4, kp = (pi & 15) << 1;
      *(unsigned*)(As + r * 32 + kp) = pack2h(av[t].x, av[t].y);
    }
#pragma unroll
    for (int t = 0; t < 4; ++t) {
      const int pi = tid + t * 256;
      const int c = pi & 63, kp = (pi >> 6) << 1;
      *(unsigned*)(Bs + c * 32 + kp) = pack2h(bv[t][0], bv[t][1]);
    }
    __syncthreads();
    const int kb = (lane >> 4) << 3;
    const v16h a  = frag_row(As + (m0 + (lane & 15)) * 32, kb);
    const v16h b0 = frag_row(Bs + (n0 + (lane & 15)) * 32, kb);
    const v16h b1 = frag_row(Bs + (n0 + 16 + (lane & 15)) * 32, kb);
    acc0 = wmma16x16x32(a, b0, acc0);
    acc1 = wmma16x16x32(a, b1, acc1);
  }
  const int nn = lane & 15, mb = m0 + ((lane >> 4) << 3);
  const int g0 = nblk + n0 + nn, g1 = g0 + 16;
  const float bc0 = bias_c[g0 < N ? g0 : N - 1];
  const float bc1 = bias_c[g1 < N ? g1 : N - 1];
#pragma unroll
  for (int r = 0; r < 8; ++r) {
    const int gm = mblk + mb + r;
    if (gm >= M) continue;
    const float br = bias_r[gm];
    if (g0 < N) Co[(long)gm * ldc + g0] = acc0[r] + br + bc0;
    if (g1 < N) Co[(long)gm * ldc + g1] = acc1[r] + br + bc1;
  }
}

// ---------------------------------------------------------------------------
// Implicit-GEMM conv (1,KH,KW) stride/pad on NCFHW fp32; KH/KW compile-time
// so the K-index decomposition is strength-reduced (no runtime division).
// W: [Cout][Cin*KH*KW] (native layout), X: [Cin][F][H][W], Y: [Cout][F][Ho][Wo]
// ---------------------------------------------------------------------------
template <int KH, int KW>
__global__ void sc_conv_igemm(const float* __restrict__ Wt, const float* __restrict__ X,
                              const float* __restrict__ bias, float* __restrict__ Y,
                              int Cin, int Cout, int F, int H, int Wd,
                              int Ho, int Wo, int stride, int pad) {
  __shared__ __align__(16) f16 As[64 * 32];
  __shared__ __align__(16) f16 Bs[64 * 32];  // transposed: [col][K]
  constexpr int khw = KH * KW;
  const int KK = Cin * khw;
  const int HoWo = Ho * Wo;
  const int N = F * HoWo;
  const int tid = threadIdx.x, lane = tid & 31, wave = tid >> 5;
  const int mblk = blockIdx.y * 64, nblk = blockIdx.x * 64;
  const int m0 = (wave & 3) * 16, n0 = (wave >> 2) * 32;
  v8f acc0 = zero8(), acc1 = zero8();
  for (int k0 = 0; k0 < KK; k0 += 32) {
    float2 av[4];
    float  bv[4][2];
#pragma unroll
    for (int t = 0; t < 4; ++t) {
      const int pi = tid + t * 256;
      const int r = pi >> 4, kp = (pi & 15) << 1;
      const int gm = mblk + r;
      const int gmc = gm < Cout ? gm : Cout - 1;
      const float2 val = *(const float2*)(Wt + (long)gmc * KK + (k0 + kp));
      av[t].x = gm < Cout ? val.x : 0.0f;
      av[t].y = gm < Cout ? val.y : 0.0f;
    }
#pragma unroll
    for (int t = 0; t < 4; ++t) {
      const int pi = tid + t * 256;
      const int c = pi & 63, kp = (pi >> 6) << 1;
      const int gn0 = nblk + c;
      const int gn = gn0 < N ? gn0 : N - 1;
      const int f = gn / HoWo, rem = gn - f * HoWo;
      const int oh = rem / Wo, ow = rem - oh * Wo;
#pragma unroll
      for (int u = 0; u < 2; ++u) {
        const int gk = k0 + kp + u;
        const int ci = gk / khw, t2 = gk - ci * khw;   // khw is constexpr
        const int ky = t2 / KW, kx = t2 - ky * KW;     // KW is constexpr
        const int ih = oh * stride - pad + ky;
        const int iw = ow * stride - pad + kx;
        const bool ok = (gn0 < N) & ((unsigned)ih < (unsigned)H) & ((unsigned)iw < (unsigned)Wd);
        const int ihc = ih < 0 ? 0 : (ih >= H ? H - 1 : ih);
        const int iwc = iw < 0 ? 0 : (iw >= Wd ? Wd - 1 : iw);
        const float v = X[(((long)ci * F + f) * H + ihc) * Wd + iwc];
        bv[t][u] = ok ? v : 0.0f;
      }
    }
    __syncthreads();
#pragma unroll
    for (int t = 0; t < 4; ++t) {
      const int pi = tid + t * 256;
      const int r = pi >> 4, kp = (pi & 15) << 1;
      *(unsigned*)(As + r * 32 + kp) = pack2h(av[t].x, av[t].y);
    }
#pragma unroll
    for (int t = 0; t < 4; ++t) {
      const int pi = tid + t * 256;
      const int c = pi & 63, kp = (pi >> 6) << 1;
      *(unsigned*)(Bs + c * 32 + kp) = pack2h(bv[t][0], bv[t][1]);
    }
    __syncthreads();
    const int kb = (lane >> 4) << 3;
    const v16h a  = frag_row(As + (m0 + (lane & 15)) * 32, kb);
    const v16h b0 = frag_row(Bs + (n0 + (lane & 15)) * 32, kb);
    const v16h b1 = frag_row(Bs + (n0 + 16 + (lane & 15)) * 32, kb);
    acc0 = wmma16x16x32(a, b0, acc0);
    acc1 = wmma16x16x32(a, b1, acc1);
  }
  const int nn = lane & 15, mb = m0 + ((lane >> 4) << 3);
#pragma unroll
  for (int r = 0; r < 8; ++r) {
    const int gm = mblk + mb + r;
    if (gm >= Cout) continue;
    const float bb = bias[gm];
    int g = nblk + n0 + nn;
#pragma unroll
    for (int half = 0; half < 2; ++half, g += 16) {
      if (g < N) {
        const int f = g / HoWo, rem = g - f * HoWo;
        const int oh = rem / Wo, ow = rem - oh * Wo;
        Y[(((long)gm * F + f) * Ho + oh) * Wo + ow] = (half ? acc1[r] : acc0[r]) + bb;
      }
    }
  }
}

// ---------------------------------------------------------------------------
// ConvTranspose3d (1,4,4) stride (1,2,2) pad (0,1,1) as 4 output-parity GEMMs.
// Wt: torch layout [Cin][Cout][4][4]. grid.z = parity (py,px).
// For output parity p, valid taps are kh = ((p+1)&1) + 2*ty, ih = (oh+1-kh)/2
// -> dense 2x2-tap conv, K = Cin*4.
// ---------------------------------------------------------------------------
__global__ void sc_convt_igemm(const float* __restrict__ Wt, const float* __restrict__ X,
                               const float* __restrict__ bias, float* __restrict__ Y,
                               int Cin, int Cout, int F, int H, int Wd) {
  __shared__ __align__(16) f16 As[64 * 32];
  __shared__ __align__(16) f16 Bs[64 * 32];  // transposed: [col][K]
  const int py = (int)blockIdx.z >> 1, px = (int)blockIdx.z & 1;
  const int khb = (py + 1) & 1, kwb = (px + 1) & 1;
  const int KK = Cin * 4;
  const int HW = H * Wd;
  const int N = F * HW;
  const int Ho = 2 * H, Wo = 2 * Wd;
  const int tid = threadIdx.x, lane = tid & 31, wave = tid >> 5;
  const int mblk = blockIdx.y * 64, nblk = blockIdx.x * 64;
  const int m0 = (wave & 3) * 16, n0 = (wave >> 2) * 32;
  v8f acc0 = zero8(), acc1 = zero8();
  for (int k0 = 0; k0 < KK; k0 += 32) {
    float av[4][2];
    float bv[4][2];
#pragma unroll
    for (int t = 0; t < 4; ++t) {
      const int pi = tid + t * 256;
      const int r = pi >> 4, kp = (pi & 15) << 1;
      const int gm0 = mblk + r;
      const int gm = gm0 < Cout ? gm0 : Cout - 1;
#pragma unroll
      for (int u = 0; u < 2; ++u) {
        const int gk = k0 + kp + u;
        const int ci = gk >> 2, tt = gk & 3;
        const int kh = khb + 2 * (tt >> 1);
        const int kw = kwb + 2 * (tt & 1);
        const float v = Wt[(((long)ci * Cout + gm) << 4) + kh * 4 + kw];
        av[t][u] = gm0 < Cout ? v : 0.0f;
      }
    }
#pragma unroll
    for (int t = 0; t < 4; ++t) {
      const int pi = tid + t * 256;
      const int c = pi & 63, kp = (pi >> 6) << 1;
      const int gn0 = nblk + c;
      const int gn = gn0 < N ? gn0 : N - 1;
      const int f = gn / HW, rem = gn - f * HW;
      const int ohh = rem / Wd, oww = rem - ohh * Wd;
      const int oh = 2 * ohh + py, ow = 2 * oww + px;
#pragma unroll
      for (int u = 0; u < 2; ++u) {
        const int gk = k0 + kp + u;
        const int ci = gk >> 2, tt = gk & 3;
        const int kh = khb + 2 * (tt >> 1);
        const int kw = kwb + 2 * (tt & 1);
        const int ih = (oh + 1 - kh) >> 1;  // numerator always even
        const int iw = (ow + 1 - kw) >> 1;
        const bool ok = (gn0 < N) & ((unsigned)ih < (unsigned)H) & ((unsigned)iw < (unsigned)Wd);
        const int ihc = ih < 0 ? 0 : (ih >= H ? H - 1 : ih);
        const int iwc = iw < 0 ? 0 : (iw >= Wd ? Wd - 1 : iw);
        const float v = X[(((long)ci * F + f) * H + ihc) * Wd + iwc];
        bv[t][u] = ok ? v : 0.0f;
      }
    }
    __syncthreads();
#pragma unroll
    for (int t = 0; t < 4; ++t) {
      const int pi = tid + t * 256;
      const int r = pi >> 4, kp = (pi & 15) << 1;
      *(unsigned*)(As + r * 32 + kp) = pack2h(av[t][0], av[t][1]);
    }
#pragma unroll
    for (int t = 0; t < 4; ++t) {
      const int pi = tid + t * 256;
      const int c = pi & 63, kp = (pi >> 6) << 1;
      *(unsigned*)(Bs + c * 32 + kp) = pack2h(bv[t][0], bv[t][1]);
    }
    __syncthreads();
    const int kb = (lane >> 4) << 3;
    const v16h a  = frag_row(As + (m0 + (lane & 15)) * 32, kb);
    const v16h b0 = frag_row(Bs + (n0 + (lane & 15)) * 32, kb);
    const v16h b1 = frag_row(Bs + (n0 + 16 + (lane & 15)) * 32, kb);
    acc0 = wmma16x16x32(a, b0, acc0);
    acc1 = wmma16x16x32(a, b1, acc1);
  }
  const int nn = lane & 15, mb = m0 + ((lane >> 4) << 3);
#pragma unroll
  for (int r = 0; r < 8; ++r) {
    const int gm = mblk + mb + r;
    if (gm >= Cout) continue;
    const float bb = bias[gm];
    int g = nblk + n0 + nn;
#pragma unroll
    for (int half = 0; half < 2; ++half, g += 16) {
      if (g < N) {
        const int f = g / HW, rem = g - f * HW;
        const int ohh = rem / Wd, oww = rem - ohh * Wd;
        const int oh = 2 * ohh + py, ow = 2 * oww + px;
        Y[(((long)gm * F + f) * Ho + oh) * Wo + ow] = (half ? acc1[r] : acc0[r]) + bb;
      }
    }
  }
}

// --------------------------- norms / elementwise ---------------------------

__global__ void sc_gn_stats(const float* __restrict__ X, float* __restrict__ stats,
                            int cg, long spatial) {
  const int g = blockIdx.x;
  const long nelem = (long)cg * spatial;
  const float* base = X + (long)g * nelem;
  float s = 0.0f, ss = 0.0f;
  for (long i = threadIdx.x; i < nelem; i += blockDim.x) {
    const float v = base[i];
    s += v; ss += v * v;
  }
  __shared__ float rs[256], rq[256];
  rs[threadIdx.x] = s; rq[threadIdx.x] = ss;
  __syncthreads();
  for (int st = 128; st > 0; st >>= 1) {
    if ((int)threadIdx.x < st) {
      rs[threadIdx.x] += rs[threadIdx.x + st];
      rq[threadIdx.x] += rq[threadIdx.x + st];
    }
    __syncthreads();
  }
  if (threadIdx.x == 0) {
    const float inv = 1.0f / (float)nelem;
    const float m = rs[0] * inv;
    const float var = rq[0] * inv - m * m;
    stats[2 * g] = m;
    stats[2 * g + 1] = rsqrtf(var + 1e-5f);
  }
}

// y = GN(x)*s+b, optional FiLM (y*(scale+1)+shift, film=[scale|shift]), optional SiLU
__global__ void sc_gn_apply(float* __restrict__ X, const float* __restrict__ stats,
                            const float* __restrict__ s, const float* __restrict__ b,
                            const float* __restrict__ film,
                            int C, int G, int spatial, int do_silu) {
  const int i = blockIdx.x * blockDim.x + threadIdx.x;
  if (i >= C * spatial) return;
  const int c = i / spatial;
  const int g = c / (C / G);
  float v = (X[i] - stats[2 * g]) * stats[2 * g + 1] * s[c] + b[c];
  if (film) v = v * (film[c] + 1.0f) + film[C + c];
  if (do_silu) v = siluf(v);
  X[i] = v;
}

// channel LayerNorm over dim C per (f,h,w); scale-only
__global__ void sc_chan_ln(const float* __restrict__ X, const float* __restrict__ g,
                           float* __restrict__ Y, int C, int spatial) {
  const int n = blockIdx.x * blockDim.x + threadIdx.x;
  if (n >= spatial) return;
  float s = 0.0f, ss = 0.0f;
  for (int c = 0; c < C; ++c) {
    const float v = X[(long)c * spatial + n];
    s += v; ss += v * v;
  }
  const float m = s / (float)C;
  const float r = rsqrtf(ss / (float)C - m * m + 1e-5f);
  for (int c = 0; c < C; ++c)
    Y[(long)c * spatial + n] = (X[(long)c * spatial + n] - m) * r * g[c];
}

__global__ void sc_time_mlp(const float* __restrict__ t, const float* __restrict__ w,
                            const float* __restrict__ b, float* __restrict__ out,
                            int Tin, int D2) {
  const int j = blockIdx.x * blockDim.x + threadIdx.x;
  if (j >= D2) return;
  float acc = b[j];
  for (int i = 0; i < Tin; ++i) acc += siluf(t[i]) * w[(long)i * D2 + j];
  out[j] = acc;
}

__global__ void sc_add2(float* __restrict__ o, const float* __restrict__ a,
                        const float* __restrict__ b, int n) {
  const int i = blockIdx.x * blockDim.x + threadIdx.x;
  if (i < n) o[i] = a[i] + b[i];
}

// (c,f,h,w) -> tokens [(l*F+f)][c], l = h*W+w
__global__ void sc_perm_cfhw_to_lfc(const float* __restrict__ X, float* __restrict__ Y,
                                    int C, int F, int HW) {
  const int idx = blockIdx.x * blockDim.x + threadIdx.x;
  if (idx >= C * F * HW) return;
  const int c = idx % C;
  const int f = (idx / C) % F;
  const int l = idx / (C * F);
  Y[idx] = X[((long)c * F + f) * HW + l];
}

// tokens back to (c,f,h,w), fused residual add
__global__ void sc_perm_lfc_to_cfhw_add(const float* __restrict__ Y, float* __restrict__ X,
                                        int C, int F, int HW) {
  const int idx = blockIdx.x * blockDim.x + threadIdx.x;
  if (idx >= C * F * HW) return;
  const int l = idx % HW;
  const int f = (idx / HW) % F;
  const int c = idx / (HW * F);
  X[idx] += Y[((long)l * F + f) * C + c];
}

// spatial linear attention core: per (frame, head) 32x32 softmax attention
// qkv: [F][96][n] (rows 0-31 q, 32-63 k, 64-95 v); outb: [F][32][n], token = head*32+m
__global__ void sc_spatial_core(const float* __restrict__ qkv, float* __restrict__ outb, int n) {
  const int fr = blockIdx.x, hd = blockIdx.y;
  __shared__ float sm[32][33];
  __shared__ float ex[32][33];
  const float* q = qkv + (long)fr * 96 * n;
  const float* k = q + 32 * n;
  const float* v = q + 64 * n;
  const int m = threadIdx.y, j = threadIdx.x;
  const int tm = hd * 32 + m, tj = hd * 32 + j;
  float s = 0.0f;
  for (int d = 0; d < 32; ++d) s += q[d * n + tm] * k[d * n + tj];
  sm[m][j] = s * 0.1767766952966369f;  // 1/sqrt(a), a=32
  __syncthreads();
  float mx = -1e30f;
  for (int t = 0; t < 32; ++t) mx = fmaxf(mx, sm[m][t]);
  const float e = __expf(sm[m][j] - mx);
  ex[m][j] = e;
  __syncthreads();
  float se = 0.0f;
  for (int t = 0; t < 32; ++t) se += ex[m][t];
  __syncthreads();
  sm[m][j] = e / se;                   // normalized attention
  __syncthreads();
  float o = 0.0f;                      // out[m][d], d = j
  for (int t = 0; t < 32; ++t) o += sm[m][t] * v[j * n + hd * 32 + t];
  outb[(long)fr * 32 * n + (long)j * n + tm] = o;
}

// temporal attention core (n_heads=1, F=8, rel pos clip +-2); q/k/v/o: [(l*F+f)][C]
__global__ void sc_temp_core(const float* __restrict__ q, const float* __restrict__ k,
                             const float* __restrict__ v, const float* __restrict__ relk,
                             const float* __restrict__ relv, float* __restrict__ o,
                             int L, int Fr, int C) {
  const int idx = blockIdx.x * blockDim.x + threadIdx.x;
  if (idx >= L * Fr) return;
  const int l = idx / Fr, qi = idx % Fr;
  const float* qrow = q + ((long)l * Fr + qi) * C;
  float w[8];
  float mx = -1e30f;
  const float inv = rsqrtf((float)C);
  for (int ki = 0; ki < Fr; ++ki) {
    int d = ki - qi; d = d < -2 ? -2 : (d > 2 ? 2 : d); d += 2;
    const float* krow = k + ((long)l * Fr + ki) * C;
    const float* rk = relk + (long)d * C;
    float s = 0.0f;
    for (int c = 0; c < C; ++c) s += qrow[c] * (krow[c] + rk[c]);
    s *= inv;
    w[ki] = s;
    mx = fmaxf(mx, s);
  }
  float se = 0.0f;
  for (int ki = 0; ki < Fr; ++ki) { w[ki] = __expf(w[ki] - mx); se += w[ki]; }
  const float rs = 1.0f / se;
  float* orow = o + ((long)l * Fr + qi) * C;
  for (int c = 0; c < C; ++c) {
    float acc = 0.0f;
    for (int ki = 0; ki < Fr; ++ki) {
      int d = ki - qi; d = d < -2 ? -2 : (d > 2 ? 2 : d); d += 2;
      acc += w[ki] * (v[((long)l * Fr + ki) * C + c] + relv[(long)d * C + c]);
    }
    orow[c] = acc * rs;
  }
}

// ------------------------------- host side --------------------------------

struct RBp {
  const float *mlp_w, *mlp_b, *w1, *b1, *gn1_s, *gn1_b, *w2, *b2, *gn2_s, *gn2_b, *rw, *rb_;
  bool has_res;
};
struct SAp { const float *ln_g, *qkv_w, *out_w, *out_b; };
struct TAp { const float *ln_g, *wq, *bq, *wk, *bk, *wv, *bv, *wo, *bo, *rel_k, *rel_v; };

extern "C" void kernel_launch(void* const* d_in, const int* in_sizes, int n_in,
                              void* d_out, int out_size, void* d_ws, size_t ws_size,
                              hipStream_t stream) {
  (void)in_sizes; (void)n_in; (void)out_size; (void)ws_size;

  const float* xin = (const float*)d_in[0];       // (1,64,8,64,64)
  const float* tvec = (const float*)d_in[1];      // (1,256)
  // d_in[2] time_rel_pos_bias, d_in[3] focus_present_mask: unused by reference.

  int pi = 4;
  auto nxt = [&]() -> const float* { return (const float*)d_in[pi++]; };
  auto read_rb = [&](bool res) {
    RBp r{};
    r.mlp_w = nxt(); r.mlp_b = nxt(); r.w1 = nxt(); r.b1 = nxt();
    r.gn1_s = nxt(); r.gn1_b = nxt(); r.w2 = nxt(); r.b2 = nxt();
    r.gn2_s = nxt(); r.gn2_b = nxt();
    r.has_res = res;
    if (res) { r.rw = nxt(); r.rb_ = nxt(); }
    return r;
  };
  auto read_sa = [&]() { SAp s{}; s.ln_g = nxt(); s.qkv_w = nxt(); s.out_w = nxt(); s.out_b = nxt(); return s; };
  auto read_ta = [&]() {
    TAp t{};
    t.ln_g = nxt(); t.wq = nxt(); t.bq = nxt(); t.wk = nxt(); t.bk = nxt();
    t.wv = nxt(); t.bv = nxt(); t.wo = nxt(); t.bo = nxt(); t.rel_k = nxt(); t.rel_v = nxt();
    return t;
  };

  RBp d10_rb1 = read_rb(true),  d10_rb2 = read_rb(false);
  const float* d10_dsw = nxt(); const float* d10_dsb = nxt();
  RBp d11_rb1 = read_rb(true),  d11_rb2 = read_rb(false);
  const float* d11_dsw = nxt(); const float* d11_dsb = nxt();
  RBp d20_rb1 = read_rb(true),  d20_rb2 = read_rb(false);
  SAp d20_sa = read_sa();       TAp d20_ta = read_ta();
  RBp mid_rb1 = read_rb(false), mid_rb2 = read_rb(false);
  SAp mid_sa = read_sa();       TAp mid_ta = read_ta();
  RBp u20_rb1 = read_rb(true),  u20_rb2 = read_rb(false);
  SAp u20_sa = read_sa();       TAp u20_ta = read_ta();
  const float* u20_usw = nxt(); const float* u20_usb = nxt();
  RBp u10_rb1 = read_rb(true),  u10_rb2 = read_rb(false);
  const float* u10_usw = nxt(); const float* u10_usb = nxt();
  RBp u11_rb1 = read_rb(true),  u11_rb2 = read_rb(false);
  const float* u11_usw = nxt(); const float* u11_usb = nxt();

  // ---- workspace carve (floats) ----
  size_t off = 0;
  auto alloc = [&](size_t nfl) -> float* {
    float* p = (float*)((char*)d_ws + off);
    off += (nfl * sizeof(float) + 255) & ~(size_t)255;
    return p;
  };
  float* zbuf  = alloc(4096);     // zeroed: dummy row/col bias
  float* temb  = alloc(2048);
  float* stats = alloc(64);
  float* tmpH  = alloc(4194304);
  float* tmpH2 = alloc(4194304);
  float* tmpR  = alloc(4194304);
  float* bufX  = alloc(8388608);
  float* bufY  = alloc(8388608);
  float* skip0 = alloc(4194304);
  float* skip1 = alloc(2097152);
  float* skip2 = alloc(1048576);
  float* xibuf = alloc(1048576);
  float* xperm = alloc(1048576);
  float* tq    = alloc(1048576);
  float* tk    = alloc(1048576);
  float* tv    = alloc(1048576);
  float* tob   = alloc(1048576);
  float* tyb   = alloc(1048576);
  float* qkvb  = alloc(196608);
  float* soutb = alloc(65536);
  float* ybuf  = alloc(1048576);

  hipMemsetAsync(zbuf, 0, 4096 * sizeof(float), stream);  // graph-capturable

  const int F = 8;

  auto gemm = [&](const float* A, long lda, long sA, const float* B, long ldb, long sB,
                  float* Cp, long ldc, long sC,
                  const float* bias_r, const float* bias_c,
                  int M, int N, int K, int batch) {
    dim3 g(cdiv_h(N, 64), cdiv_h(M, 64), batch);
    sc_gemm_wmma<<<g, 256, 0, stream>>>(A, lda, sA, B, ldb, sB, Cp, ldc, sC,
                                        bias_r ? bias_r : zbuf, bias_c ? bias_c : zbuf,
                                        M, N, K);
  };
  auto conv = [&](const float* w, const float* x, const float* b, float* y,
                  int Cin, int Cout, int H, int W, int KH, int KW, int stride, int pad) {
    const int Ho = (H + 2 * pad - KH) / stride + 1;
    const int Wo = (W + 2 * pad - KW) / stride + 1;
    dim3 g(cdiv_h(F * Ho * Wo, 64), cdiv_h(Cout, 64), 1);
    if (KH == 3)
      sc_conv_igemm<3, 3><<<g, 256, 0, stream>>>(w, x, b, y, Cin, Cout, F, H, W, Ho, Wo, stride, pad);
    else if (KH == 4)
      sc_conv_igemm<4, 4><<<g, 256, 0, stream>>>(w, x, b, y, Cin, Cout, F, H, W, Ho, Wo, stride, pad);
    else
      sc_conv_igemm<1, 1><<<g, 256, 0, stream>>>(w, x, b, y, Cin, Cout, F, H, W, Ho, Wo, stride, pad);
  };
  auto convt = [&](const float* w, const float* x, const float* b, float* y,
                   int Cio, int H, int W) {
    dim3 g(cdiv_h(F * H * W, 64), cdiv_h(Cio, 64), 4);
    sc_convt_igemm<<<g, 256, 0, stream>>>(w, x, b, y, Cio, Cio, F, H, W);
  };
  auto resnet = [&](const RBp& p, const float* xi, float* xo, int Cin, int Cout, int H, int W) {
    const int S = F * H * W;
    sc_time_mlp<<<cdiv_h(2 * Cout, 256), 256, 0, stream>>>(tvec, p.mlp_w, p.mlp_b, temb, 256, 2 * Cout);
    conv(p.w1, xi, p.b1, tmpH, Cin, Cout, H, W, 3, 3, 1, 1);
    sc_gn_stats<<<8, 256, 0, stream>>>(tmpH, stats, Cout / 8, (long)S);
    sc_gn_apply<<<cdiv_h(Cout * S, 256), 256, 0, stream>>>(tmpH, stats, p.gn1_s, p.gn1_b, temb, Cout, 8, S, 1);
    conv(p.w2, tmpH, p.b2, tmpH2, Cout, Cout, H, W, 3, 3, 1, 1);
    sc_gn_stats<<<8, 256, 0, stream>>>(tmpH2, stats, Cout / 8, (long)S);
    sc_gn_apply<<<cdiv_h(Cout * S, 256), 256, 0, stream>>>(tmpH2, stats, p.gn2_s, p.gn2_b, nullptr, Cout, 8, S, 1);
    const float* res = xi;
    if (p.has_res) { conv(p.rw, xi, p.rb_, tmpR, Cin, Cout, H, W, 1, 1, 1, 0); res = tmpR; }
    sc_add2<<<cdiv_h(Cout * S, 256), 256, 0, stream>>>(xo, tmpH2, res, Cout * S);
  };
  auto spatial = [&](const SAp& p, float* x, int C, int HW) {
    const int S = F * HW;
    sc_chan_ln<<<cdiv_h(S, 256), 256, 0, stream>>>(x, p.ln_g, xibuf, C, S);
    gemm(p.qkv_w, C, 0, xibuf, (long)S, (long)HW, qkvb, (long)HW, (long)96 * HW,
         nullptr, nullptr, 96, HW, C, F);
    sc_spatial_core<<<dim3(F, 8), dim3(32, 32), 0, stream>>>(qkvb, soutb, HW);
    gemm(p.out_w, 32, 0, soutb, (long)HW, (long)32 * HW, ybuf, (long)S, (long)HW,
         p.out_b, nullptr, C, HW, 32, F);
    sc_add2<<<cdiv_h(C * S, 256), 256, 0, stream>>>(x, x, ybuf, C * S);
  };
  auto temporal = [&](const TAp& p, float* x, int C, int HW) {
    const int S = F * HW;
    const int T = HW * F;
    sc_chan_ln<<<cdiv_h(S, 256), 256, 0, stream>>>(x, p.ln_g, xibuf, C, S);
    sc_perm_cfhw_to_lfc<<<cdiv_h(T * C, 256), 256, 0, stream>>>(xibuf, xperm, C, F, HW);
    gemm(xperm, C, 0, p.wq, C, 0, tq, C, 0, nullptr, p.bq, T, C, C, 1);
    gemm(xperm, C, 0, p.wk, C, 0, tk, C, 0, nullptr, p.bk, T, C, C, 1);
    gemm(xperm, C, 0, p.wv, C, 0, tv, C, 0, nullptr, p.bv, T, C, C, 1);
    sc_temp_core<<<cdiv_h(T, 256), 256, 0, stream>>>(tq, tk, tv, p.rel_k, p.rel_v, tob, HW, F, C);
    gemm(tob, C, 0, p.wo, C, 0, tyb, C, 0, nullptr, p.bo, T, C, C, 1);
    sc_perm_lfc_to_cfhw_add<<<cdiv_h(C * S, 256), 256, 0, stream>>>(tyb, x, C, F, HW);
  };
  auto concat2 = [&](float* dst, const float* a, long na, const float* b, long nb) {
    hipMemcpyAsync(dst, a, na * sizeof(float), hipMemcpyDeviceToDevice, stream);
    hipMemcpyAsync(dst + na, b, nb * sizeof(float), hipMemcpyDeviceToDevice, stream);
  };

  // ------------------------------ pipeline ------------------------------
  // down1_0 @64x64
  resnet(d10_rb1, xin, bufX, 64, 128, 64, 64);
  resnet(d10_rb2, bufX, skip0, 128, 128, 64, 64);
  conv(d10_dsw, skip0, d10_dsb, bufX, 128, 128, 64, 64, 4, 4, 2, 1);  // -> 32x32
  // down1_1 @32x32
  resnet(d11_rb1, bufX, bufY, 128, 256, 32, 32);
  resnet(d11_rb2, bufY, skip1, 256, 256, 32, 32);
  conv(d11_dsw, skip1, d11_dsb, bufX, 256, 256, 32, 32, 4, 4, 2, 1);  // -> 16x16
  // down2_0 @16x16
  resnet(d20_rb1, bufX, bufY, 256, 512, 16, 16);
  resnet(d20_rb2, bufY, bufX, 512, 512, 16, 16);
  spatial(d20_sa, bufX, 512, 256);
  temporal(d20_ta, bufX, 512, 256);
  hipMemcpyAsync(skip2, bufX, (size_t)512 * 2048 * sizeof(float), hipMemcpyDeviceToDevice, stream);
  // mid
  resnet(mid_rb1, bufX, bufY, 512, 512, 16, 16);
  spatial(mid_sa, bufY, 512, 256);
  temporal(mid_ta, bufY, 512, 256);
  resnet(mid_rb2, bufY, bufX, 512, 512, 16, 16);
  // up2_0
  concat2(bufY, bufX, (long)512 * 2048, skip2, (long)512 * 2048);
  resnet(u20_rb1, bufY, bufX, 1024, 256, 16, 16);
  resnet(u20_rb2, bufX, bufY, 256, 256, 16, 16);
  spatial(u20_sa, bufY, 256, 256);
  temporal(u20_ta, bufY, 256, 256);
  convt(u20_usw, bufY, u20_usb, bufX, 256, 16, 16);                   // -> 256 @32x32
  // up1_0
  concat2(bufY, bufX, (long)256 * 8192, skip1, (long)256 * 8192);
  resnet(u10_rb1, bufY, bufX, 512, 128, 32, 32);
  resnet(u10_rb2, bufX, bufY, 128, 128, 32, 32);
  convt(u10_usw, bufY, u10_usb, bufX, 128, 32, 32);                   // -> 128 @64x64
  // up1_1
  concat2(bufY, bufX, (long)128 * 32768, skip0, (long)128 * 32768);
  resnet(u11_rb1, bufY, bufX, 256, 64, 64, 64);
  resnet(u11_rb2, bufX, bufY, 64, 64, 64, 64);
  convt(u11_usw, bufY, u11_usb, (float*)d_out, 64, 64, 64);           // -> 64 @128x128 (output)
}
